// MoeForward_94489280667
// MI455X (gfx1250) — matrix operand order repaired and checked by reading, not compiled, not run on hardware
//
#include <hip/hip_runtime.h>

// ---------- types ----------
typedef __bf16 bf16_t;
typedef __attribute__((ext_vector_type(16))) __bf16 v16bf;
typedef __attribute__((ext_vector_type(8)))  __bf16 v8bf;
typedef __attribute__((ext_vector_type(8)))  float  v8f;

#define T_TOK 2048
#define DD    512
#define FF    1024
#define EE    8

static __device__ __forceinline__ bf16_t f32_to_bf16(float f) {
    unsigned u = __builtin_bit_cast(unsigned, f);
    unsigned r = (u + 0x7FFFu + ((u >> 16) & 1u)) >> 16;   // round-to-nearest-even
    unsigned short s = (unsigned short)r;
    return __builtin_bit_cast(bf16_t, s);
}

// ---------- kernel: zero output ----------
__global__ void zero_f32(float* __restrict__ p, int n) {
    int i = blockIdx.x * blockDim.x + threadIdx.x;
    if (i < n) p[i] = 0.0f;
}

// ---------- kernel: fp32 -> bf16 with optional per-expert transpose ----------
// src layout [nE][R][C] -> dst layout [nE][C][R]   (nE=1,R=1 => plain convert)
__global__ void convert_transpose(const float* __restrict__ src,
                                  bf16_t* __restrict__ dst,
                                  int nE, int R, int C) {
    size_t idx = (size_t)blockIdx.x * blockDim.x + threadIdx.x;
    size_t total = (size_t)nE * R * C;
    if (idx >= total) return;
    int rc = R * C;
    int e  = (int)(idx / rc);
    int rem = (int)(idx % rc);
    int r  = rem / C;
    int c  = rem % C;
    dst[(size_t)e * rc + (size_t)c * R + r] = f32_to_bf16(src[idx]);
}

// ---------- kernel: router (softmax + top-2 + renorm -> dense combine [T,E]) ----------
__global__ void router_kernel(const float* __restrict__ x,
                              const float* __restrict__ rw,   // [D][E]
                              float* __restrict__ combine) {  // [T][E]
    int t = blockIdx.x * blockDim.x + threadIdx.x;
    if (t >= T_TOK) return;
    float acc[EE];
#pragma unroll
    for (int e = 0; e < EE; ++e) acc[e] = 0.0f;
    const float* xr = x + (size_t)t * DD;
    for (int d = 0; d < DD; ++d) {
        float xv = xr[d];
        const float* r = rw + (size_t)d * EE;
#pragma unroll
        for (int e = 0; e < EE; ++e) acc[e] += xv * r[e];
    }
    float m = acc[0];
#pragma unroll
    for (int e = 1; e < EE; ++e) m = fmaxf(m, acc[e]);
    float p[EE]; float sum = 0.0f;
#pragma unroll
    for (int e = 0; e < EE; ++e) { p[e] = __expf(acc[e] - m); sum += p[e]; }
    float inv = 1.0f / sum;
#pragma unroll
    for (int e = 0; e < EE; ++e) p[e] *= inv;
    int e1 = 0; float v1 = p[0];
#pragma unroll
    for (int e = 1; e < EE; ++e) if (p[e] > v1) { v1 = p[e]; e1 = e; }
    int e2 = -1; float v2 = -1.0f;
#pragma unroll
    for (int e = 0; e < EE; ++e) if (e != e1 && p[e] > v2) { v2 = p[e]; e2 = e; }
    float wsum = v1 + v2;
#pragma unroll
    for (int e = 0; e < EE; ++e) {
        float c = 0.0f;
        if (e == e1) c = v1 / wsum;
        else if (e == e2) c = v2 / wsum;
        combine[(size_t)t * EE + e] = c;
    }
}

// ---------- main MoE kernel: one (16-token tile, expert) per block ----------
// 256 threads = 8 wave32s. GEMM1 (gate/up, K=512) -> silu*up*combine -> LDS(bf16)
// -> GEMM2 (down, K=1024) -> atomic add fp32 into out.
__global__ __launch_bounds__(256) void moe_main(
    const bf16_t* __restrict__ x_bf,     // [T][D]
    const bf16_t* __restrict__ w1t,      // [E][F][D]  (transposed gate)
    const bf16_t* __restrict__ w3t,      // [E][F][D]  (transposed up)
    const bf16_t* __restrict__ w2t,      // [E][D][F]  (transposed down)
    const float*  __restrict__ combine,  // [T][E]
    float*        __restrict__ out)      // [T][D]
{
    __shared__ bf16_t sh_h[16][FF + 8];  // +8 bf16 pad (16B) -> conflict-free ds_load_b128
    __shared__ float  sh_c[16];
    __shared__ int    sh_any;

    const int tid  = threadIdx.x;
    const int wave = tid >> 5;           // wave32
    const int lane = tid & 31;
    const int hi   = lane >> 4;          // half-wave select (K-chunk select)
    const int col  = lane & 15;          // N column / A row
    const int tile = blockIdx.x;
    const int e    = blockIdx.y;
    const int tokenBase = tile * 16;

    if (tid == 0) sh_any = 0;
    __syncthreads();
    if (tid < 16) {
        float c = combine[(size_t)(tokenBase + tid) * EE + e];
        sh_c[tid] = c;
        if (c != 0.0f) atomicOr(&sh_any, 1);
    }
    __syncthreads();
    if (sh_any == 0) return;             // no token in tile picked this expert

    const bf16_t* xrow = x_bf + (size_t)(tokenBase + col) * DD;

    // ---------------- GEMM1: gate = x@w1, up = x@w3 ----------------
    for (int j = 0; j < 8; ++j) {
        const int ft = wave * 8 + j;     // F-tile 0..63
        v8f accg = {}; v8f accu = {};
        const bf16_t* b1 = w1t + ((size_t)e * FF + ft * 16 + col) * DD + hi * 16;
        const bf16_t* b3 = w3t + ((size_t)e * FF + ft * 16 + col) * DD + hi * 16;
        for (int kk = 0; kk < DD; kk += 32) {
            // A 16x32 bf16: lane half selects 16 contiguous K starting at kk+hi*16
            v8bf alo = *(const v8bf*)(xrow + kk + hi * 16);
            v8bf ahi = *(const v8bf*)(xrow + kk + hi * 16 + 8);
            v16bf A;
#pragma unroll
            for (int i = 0; i < 8; ++i) { A[i] = alo[i]; A[i + 8] = ahi[i]; }
            // B 32x16 bf16: lane = column, 16 contiguous K starting at kk+hi*16 (transposed weights)
            v16bf B1 = *(const v16bf*)(b1 + kk);
            v16bf B3 = *(const v16bf*)(b3 + kk);
            accg = __builtin_amdgcn_wmma_f32_16x16x32_bf16(false, A, false, B1,
                                                           (short)0, accg, false, false);
            accu = __builtin_amdgcn_wmma_f32_16x16x32_bf16(false, A, false, B3,
                                                           (short)0, accu, false, false);
        }
        // epilogue: h = silu(gate)*up*combine -> LDS as bf16
#pragma unroll
        for (int i = 0; i < 8; ++i) {
            const int m = hi * 8 + i;    // C layout: vgpr i, lanes0-15 M=i, lanes16-31 M=i+8
            float g = accg[i];
            float s = g / (1.0f + __expf(-g));        // silu
            float hv = s * accu[i] * sh_c[m];
            sh_h[m][ft * 16 + col] = f32_to_bf16(hv);
        }
    }
    __syncthreads();

    // ---------------- GEMM2: out += h @ w2 ----------------
    const bf16_t* arow = &sh_h[col][0];
    for (int j = 0; j < 4; ++j) {
        const int dt = wave * 4 + j;     // D-tile 0..31
        v8f acc = {};
        const bf16_t* b2 = w2t + ((size_t)e * DD + dt * 16 + col) * FF + hi * 16;
        for (int kk = 0; kk < FF; kk += 32) {
            v8bf alo = *(const v8bf*)(arow + kk + hi * 16);
            v8bf ahi = *(const v8bf*)(arow + kk + hi * 16 + 8);
            v16bf A;
#pragma unroll
            for (int i = 0; i < 8; ++i) { A[i] = alo[i]; A[i + 8] = ahi[i]; }
            v16bf B = *(const v16bf*)(b2 + kk);
            acc = __builtin_amdgcn_wmma_f32_16x16x32_bf16(false, A, false, B,
                                                          (short)0, acc, false, false);
        }
#pragma unroll
        for (int i = 0; i < 8; ++i) {
            const int m = hi * 8 + i;
            atomicAdd(&out[(size_t)(tokenBase + m) * DD + dt * 16 + col], acc[i]);
        }
    }
}

// ---------- host launch ----------
extern "C" void kernel_launch(void* const* d_in, const int* in_sizes, int n_in,
                              void* d_out, int out_size, void* d_ws, size_t ws_size,
                              hipStream_t stream) {
    (void)in_sizes; (void)n_in; (void)out_size; (void)ws_size;
    const float* x  = (const float*)d_in[0];   // [1,2048,512]
    const float* rw = (const float*)d_in[1];   // [512,8]
    const float* w1 = (const float*)d_in[2];   // [8,512,1024] gate
    const float* w2 = (const float*)d_in[3];   // [8,1024,512] down
    const float* w3 = (const float*)d_in[4];   // [8,512,1024] up
    float* out = (float*)d_out;                // [1,2048,512]

    // workspace layout (bytes)
    char* ws = (char*)d_ws;
    const size_t xbf_off  = 0;                                  // 2048*512*2   = 2 MB
    const size_t w1t_off  = xbf_off + (size_t)T_TOK * DD * 2;   // 8*1024*512*2 = 8 MB
    const size_t w3t_off  = w1t_off + (size_t)EE * DD * FF * 2;
    const size_t w2t_off  = w3t_off + (size_t)EE * DD * FF * 2;
    const size_t comb_off = w2t_off + (size_t)EE * DD * FF * 2; // 2048*8*4 = 64 KB
    bf16_t* x_bf = (bf16_t*)(ws + xbf_off);
    bf16_t* w1t  = (bf16_t*)(ws + w1t_off);
    bf16_t* w3t  = (bf16_t*)(ws + w3t_off);
    bf16_t* w2t  = (bf16_t*)(ws + w2t_off);
    float* comb  = (float*)(ws + comb_off);

    const int nOut = T_TOK * DD;                 // 1,048,576
    zero_f32<<<(nOut + 255) / 256, 256, 0, stream>>>(out, nOut);

    // convert x (no transpose)
    convert_transpose<<<(T_TOK * DD + 255) / 256, 256, 0, stream>>>(x, x_bf, 1, 1, T_TOK * DD);
    // w1 [E][D][F] -> [E][F][D]
    const int nW = EE * DD * FF;                 // 4,194,304
    convert_transpose<<<(nW + 255) / 256, 256, 0, stream>>>(w1, w1t, EE, DD, FF);
    // w3 [E][D][F] -> [E][F][D]
    convert_transpose<<<(nW + 255) / 256, 256, 0, stream>>>(w3, w3t, EE, DD, FF);
    // w2 [E][F][D] -> [E][D][F]
    convert_transpose<<<(nW + 255) / 256, 256, 0, stream>>>(w2, w2t, EE, FF, DD);

    // router: 1 thread / token
    router_kernel<<<(T_TOK + 255) / 256, 256, 0, stream>>>(x, rw, comb);

    // main MoE: (token-tile, expert) grid
    dim3 grid(T_TOK / 16, EE);
    moe_main<<<grid, 256, 0, stream>>>(x_bf, w1t, w3t, w2t, comb, out);
}
